// BahdanauAttentionMOMA_63410897158223
// MI455X (gfx1250) — matrix operand order repaired and checked by reading, not compiled, not run on hardware
//
#include <hip/hip_runtime.h>
#include <math.h>

// ---------------- types ----------------
typedef __bf16 bf16_t;
typedef __attribute__((ext_vector_type(16))) __bf16 v16bf;
typedef __attribute__((ext_vector_type(8)))  __bf16 v8bf;
typedef __attribute__((ext_vector_type(8)))  float  v8f;

#define S_DIM 1024
#define B_DIM 32
#define E_DIM 1024
#define VD_DIM 1024
#define M_TOT (S_DIM * B_DIM)   // 32768
#define M_TILE 64
#define AROW (VD_DIM + 8)       // 1032 halves: 16B-aligned row stride for A tile
#define NCHUNK 256              // N columns per chunk (4 chunks cover E=1024)
#define KSLAB 256               // K columns per staged B slab (4 slabs cover K=1024)
#define BROW (KSLAB + 8)        // 264 halves: 16B-aligned row stride for B slab

__device__ __forceinline__ float tanh_fast(float x) {
  // branchless; __expf saturates so limits are exact (+1 / -1)
  return 1.0f - 2.0f / (__expf(2.0f * x) + 1.0f);
}

// ---------------- kernel A: normed_v = g * v / ||v|| ----------------
__global__ void norm_v_kernel(const float* __restrict__ v1, const float* __restrict__ g1,
                              const float* __restrict__ v2, const float* __restrict__ g2,
                              const float* __restrict__ v3, const float* __restrict__ g3,
                              float* __restrict__ nv /* [3][E] */) {
  __shared__ float red[256];
  const int br = blockIdx.x;
  const float* v = (br == 0) ? v1 : ((br == 1) ? v2 : v3);
  const float* g = (br == 0) ? g1 : ((br == 1) ? g2 : g3);
  const int t = threadIdx.x;
  float acc = 0.f;
  for (int e = t; e < E_DIM; e += 256) { float x = v[e]; acc += x * x; }
  red[t] = acc; __syncthreads();
  for (int w = 128; w > 0; w >>= 1) { if (t < w) red[t] += red[t + w]; __syncthreads(); }
  const float scale = g[0] * rsqrtf(red[0]);
  for (int e = t; e < E_DIM; e += 256) nv[br * E_DIM + e] = v[e] * scale;
}

// ---------------- kernel B: pqb[br][b][e] = query[b]·Wq[e] + bias[e] ----------------
__global__ void pq_kernel(const float* __restrict__ query,
                          const float* __restrict__ Wq1, const float* __restrict__ b1,
                          const float* __restrict__ Wq2, const float* __restrict__ b2,
                          const float* __restrict__ Wq3, const float* __restrict__ b3,
                          float* __restrict__ pqb /* [3][B][E] */) {
  const int blk = blockIdx.x;           // 96 blocks = 3 br * 32 b
  const int br = blk >> 5, b = blk & 31;
  const float* Wq   = (br == 0) ? Wq1 : ((br == 1) ? Wq2 : Wq3);
  const float* bias = (br == 0) ? b1  : ((br == 1) ? b2  : b3);
  __shared__ float q[1024];
  const int t = threadIdx.x;
  for (int d = t; d < 1024; d += 256) q[d] = query[b * 1024 + d];
  __syncthreads();
  for (int e = t; e < E_DIM; e += 256) {
    const float4* wr = (const float4*)(Wq + (size_t)e * 1024);
    float acc = 0.f;
#pragma unroll 4
    for (int d4 = 0; d4 < 256; ++d4) {
      float4 w = wr[d4];
      acc += w.x * q[d4 * 4 + 0] + w.y * q[d4 * 4 + 1] +
             w.z * q[d4 * 4 + 2] + w.w * q[d4 * 4 + 3];
    }
    pqb[(size_t)br * (B_DIM * E_DIM) + b * E_DIM + e] = acc + bias[e];
  }
}

// ---------------- kernel C: fused WMMA GEMM + tanh + reduce -> scores ----------------
// 512 threads (16 waves), grid = 512 M-tiles of 64 rows.
// Each wave owns a 32x32 output region (2x2 wmma tiles): 4 wmma per 8 ds_load_b128.
// LDS: A 64x1032 bf16 (129K) + B slab 256x264 bf16 (132K) + pq 32K + nv 4K + score 256B ~ 304KB
__global__ __launch_bounds__(512)
void scores_kernel(const float* __restrict__ value,
                   const float* __restrict__ Wv1, const float* __restrict__ Wv2,
                   const float* __restrict__ Wv3,
                   const float* __restrict__ pqb,   // [3][B][E]
                   const float* __restrict__ nvg,   // [3][E]
                   float* __restrict__ scores) {    // [3][M_TOT]
  extern __shared__ char smem[];
  bf16_t* As  = (bf16_t*)smem;                       // [64][AROW]
  bf16_t* Bs  = (bf16_t*)(smem + 132096);            // [256][BROW]
  float*  pqs = (float*)(smem + 132096 + 135168);    // [32][256]
  float*  nvs = pqs + 32 * 256;                      // [1024]
  float*  scs = nvs + 1024;                          // [64]

  const int tid    = threadIdx.x;
  const int lane   = tid & 31;
  const int wave   = tid >> 5;
  const int m_grp  = wave >> 3;   // 0..1  -> 32 rows each
  const int n_grp  = wave & 7;    // 0..7  -> 32 cols each
  const int laneLo = lane & 15;
  const int laneHi = lane >> 4;
  const int Mbase  = blockIdx.x * M_TILE;

  // Stage A tile once: value rows [Mbase, Mbase+64) -> bf16 (value HBM-read exactly once)
  for (int i = tid; i < M_TILE * VD_DIM; i += 512) {
    const int r = i >> 10, cc = i & 1023;
    As[r * AROW + cc] = (bf16_t)value[(size_t)(Mbase + r) * VD_DIM + cc];
  }

  union Frag { v16bf v; v8bf h[2]; };

  for (int br = 0; br < 3; ++br) {
    const float* Wv = (br == 0) ? Wv1 : ((br == 1) ? Wv2 : Wv3);
    __syncthreads();  // A staged / previous branch done with nvs & scs
    for (int i = tid; i < E_DIM; i += 512) nvs[i] = nvg[br * E_DIM + i];
    if (tid < M_TILE) scs[tid] = 0.f;

    float part[16];
#pragma unroll
    for (int i = 0; i < 16; ++i) part[i] = 0.f;

    for (int chunk = 0; chunk < 4; ++chunk) {       // N chunks of 256
      v8f cacc[2][2];
#pragma unroll
      for (int mi = 0; mi < 2; ++mi)
#pragma unroll
        for (int ni = 0; ni < 2; ++ni) cacc[mi][ni] = v8f{};

      for (int slab = 0; slab < 4; ++slab) {        // K slabs of 256
        __syncthreads();  // previous compute done before restaging Bs
        // Stage B slab: Wv rows [chunk*256, +256) x K cols [slab*256, +256) -> bf16
        for (int i = tid; i < NCHUNK * KSLAB; i += 512) {
          const int r = i >> 8, cc = i & 255;
          Bs[r * BROW + cc] =
              (bf16_t)Wv[(size_t)(chunk * NCHUNK + r) * VD_DIM + slab * KSLAB + cc];
        }
        if (slab == 0) {  // stage pq+bias for this branch/chunk: [32 b][256 n]
          for (int i = tid; i < 32 * NCHUNK; i += 512) {
            const int bb = i >> 8, nn = i & 255;
            pqs[bb * NCHUNK + nn] =
                pqb[(size_t)br * (B_DIM * E_DIM) + bb * E_DIM + chunk * NCHUNK + nn];
          }
        }
        __syncthreads();

        // Prefetch the next Wv slab into L2/WGP$ while this slab computes
        {
          const int next = chunk * 4 + slab + 1;
          if (next < 16) {
            const float* nb = Wv + (size_t)((next >> 2) * NCHUNK) * VD_DIM + (next & 3) * KSLAB;
            for (int i = tid; i < 2048; i += 512)   // 256 rows x 8 cachelines(128B)
              __builtin_prefetch(nb + (size_t)(i >> 3) * VD_DIM + (i & 7) * 32, 0, 1);
          }
        }

        // --- 2x2 wmma tiles, software-pipelined fragment loads ---
        const bf16_t* aP = As + (m_grp * 32 + laneLo) * AROW + 8 * laneHi + slab * KSLAB;
        const bf16_t* aQ = aP + 16 * AROW;
        const bf16_t* bP = Bs + (n_grp * 32 + laneLo) * BROW + 8 * laneHi;
        const bf16_t* bQ = bP + 16 * BROW;

        Frag a0[2], a1[2], b0[2], b1[2];
        a0[0].h[0] = *(const v8bf*)(aP);      a0[0].h[1] = *(const v8bf*)(aP + 16);
        a1[0].h[0] = *(const v8bf*)(aQ);      a1[0].h[1] = *(const v8bf*)(aQ + 16);
        b0[0].h[0] = *(const v8bf*)(bP);      b0[0].h[1] = *(const v8bf*)(bP + 16);
        b1[0].h[0] = *(const v8bf*)(bQ);      b1[0].h[1] = *(const v8bf*)(bQ + 16);

#pragma unroll
        for (int kk = 0; kk < KSLAB; kk += 32) {
          const int cur = (kk >> 5) & 1;
          const int nxt = cur ^ 1;
          if (kk + 32 < KSLAB) {   // issue next-step loads before current wmmas
            a0[nxt].h[0] = *(const v8bf*)(aP + kk + 32);
            a0[nxt].h[1] = *(const v8bf*)(aP + kk + 48);
            a1[nxt].h[0] = *(const v8bf*)(aQ + kk + 32);
            a1[nxt].h[1] = *(const v8bf*)(aQ + kk + 48);
            b0[nxt].h[0] = *(const v8bf*)(bP + kk + 32);
            b0[nxt].h[1] = *(const v8bf*)(bP + kk + 48);
            b1[nxt].h[0] = *(const v8bf*)(bQ + kk + 32);
            b1[nxt].h[1] = *(const v8bf*)(bQ + kk + 48);
          }
          cacc[0][0] = __builtin_amdgcn_wmma_f32_16x16x32_bf16(false, a0[cur].v, false, b0[cur].v,
                                                               (short)0, cacc[0][0], false, false);
          cacc[0][1] = __builtin_amdgcn_wmma_f32_16x16x32_bf16(false, a0[cur].v, false, b1[cur].v,
                                                               (short)0, cacc[0][1], false, false);
          cacc[1][0] = __builtin_amdgcn_wmma_f32_16x16x32_bf16(false, a1[cur].v, false, b0[cur].v,
                                                               (short)0, cacc[1][0], false, false);
          cacc[1][1] = __builtin_amdgcn_wmma_f32_16x16x32_bf16(false, a1[cur].v, false, b1[cur].v,
                                                               (short)0, cacc[1][1], false, false);
        }
      } // slab

      // epilogue: out = GEMM + pq -> tanh -> * normed_v, accumulate per-row partials
      // C layout: VGPR r -> M = r + 8*laneHi, N = laneLo (within a 16x16 tile)
#pragma unroll
      for (int mi = 0; mi < 2; ++mi) {
#pragma unroll
        for (int r = 0; r < 8; ++r) {
          const int m64  = m_grp * 32 + mi * 16 + r + 8 * laneHi;
          const int bidx = (Mbase + m64) & 31;
          float p = 0.f;
#pragma unroll
          for (int ni = 0; ni < 2; ++ni) {
            const int nC  = n_grp * 32 + ni * 16 + laneLo;
            const float val = cacc[mi][ni][r] + pqs[bidx * NCHUNK + nC];
            p += nvs[chunk * NCHUNK + nC] * tanh_fast(val);
          }
          part[mi * 8 + r] += p;
        }
      }
    } // chunk

    // reduce per-lane partials into LDS scores
#pragma unroll
    for (int mi = 0; mi < 2; ++mi)
#pragma unroll
      for (int r = 0; r < 8; ++r) {
        const int m64 = m_grp * 32 + mi * 16 + r + 8 * laneHi;
        atomicAdd(&scs[m64], part[mi * 8 + r]);
      }
    __syncthreads();
    if (tid < M_TILE) scores[(size_t)br * M_TOT + Mbase + tid] = scs[tid];
  } // branch
}

// ---------------- kernel D: softmax over s per (branch, b) ----------------
__global__ void softmax_kernel(const float* __restrict__ scores,
                               const unsigned char* __restrict__ mask,
                               float* __restrict__ attn) {
  const int blk = blockIdx.x;      // 96 = 3 br * 32 b
  const int br = blk >> 5, b = blk & 31;
  __shared__ float red[256];
  __shared__ float buf[1024];
  const int t = threadIdx.x;
  float mx = -INFINITY;
  for (int s = t; s < S_DIM; s += 256) {
    float sc = scores[(size_t)br * M_TOT + s * B_DIM + b];
    if (mask[s * B_DIM + b]) sc = -INFINITY;
    buf[s] = sc;
    mx = fmaxf(mx, sc);
  }
  red[t] = mx; __syncthreads();
  for (int w = 128; w > 0; w >>= 1) { if (t < w) red[t] = fmaxf(red[t], red[t + w]); __syncthreads(); }
  mx = red[0];
  __syncthreads();
  float sum = 0.f;
  for (int s = t; s < S_DIM; s += 256) { float e = __expf(buf[s] - mx); buf[s] = e; sum += e; }
  red[t] = sum; __syncthreads();
  for (int w = 128; w > 0; w >>= 1) { if (t < w) red[t] += red[t + w]; __syncthreads(); }
  const float inv = 1.f / red[0];
  for (int s = t; s < S_DIM; s += 256)
    attn[(size_t)br * M_TOT + s * B_DIM + b] = buf[s] * inv;
}

// ---------------- kernel E1: combine attn -> comb, write attn outputs ----------------
__global__ void combine_kernel(const float* __restrict__ attn,
                               const float* __restrict__ g1, const float* __restrict__ g2,
                               const float* __restrict__ g3,
                               float* __restrict__ comb,
                               float* __restrict__ out_a1, float* __restrict__ out_a2) {
  const int i = blockIdx.x * 256 + threadIdx.x;
  if (i < M_TOT) {
    const float cv = g1[0] * attn[i] + g2[0] * attn[M_TOT + i] + g3[0] * attn[2 * M_TOT + i];
    comb[i] = cv; out_a1[i] = cv; out_a2[i] = cv;
  }
}

// ---------------- kernel E2: context[b][d] = sum_s comb[s,b]*value[s,b,d] ----------------
__global__ void context_kernel(const float* __restrict__ value,
                               const float* __restrict__ comb,
                               float* __restrict__ ctx) {
  const int b = blockIdx.x >> 2;            // 32 b * 4 d-chunks
  const int d = (blockIdx.x & 3) * 256 + threadIdx.x;
  float acc = 0.f;
  for (int s = 0; s < S_DIM; ++s) {
    const float w = comb[(s << 5) + b];     // uniform per block -> scalar load
    acc += w * value[(size_t)(s * B_DIM + b) * VD_DIM + d];
  }
  ctx[b * VD_DIM + d] = acc;
}

// ---------------- launch ----------------
extern "C" void kernel_launch(void* const* d_in, const int* in_sizes, int n_in,
                              void* d_out, int out_size, void* d_ws, size_t ws_size,
                              hipStream_t stream) {
  const float* query = (const float*)d_in[0];
  const float* value = (const float*)d_in[1];
  const unsigned char* mask = (const unsigned char*)d_in[2];
  const float *Wq1 = (const float*)d_in[3],  *Wv1 = (const float*)d_in[4],
              *v1  = (const float*)d_in[5],  *b1  = (const float*)d_in[6],
              *g1  = (const float*)d_in[7],  *ga1 = (const float*)d_in[8];
  const float *Wq2 = (const float*)d_in[9],  *Wv2 = (const float*)d_in[10],
              *v2  = (const float*)d_in[11], *b2  = (const float*)d_in[12],
              *g2  = (const float*)d_in[13], *ga2 = (const float*)d_in[14];
  const float *Wq3 = (const float*)d_in[15], *Wv3 = (const float*)d_in[16],
              *v3  = (const float*)d_in[17], *b3  = (const float*)d_in[18],
              *g3  = (const float*)d_in[19], *ga3 = (const float*)d_in[20];

  float* out = (float*)d_out;      // [context 32768 | attn 32768 | attn 32768]

  // workspace layout
  char* ws = (char*)d_ws;
  float* nv     = (float*)(ws);                        // 3*1024
  float* pqb    = (float*)(ws + 12288);                // 3*32*1024
  float* scores = (float*)(ws + 12288 + 393216);       // 3*32768
  float* attn   = (float*)(ws + 12288 + 2 * 393216);   // 3*32768
  float* comb   = (float*)(ws + 12288 + 3 * 393216);   // 32768

  norm_v_kernel<<<3, 256, 0, stream>>>(v1, g1, v2, g2, v3, g3, nv);
  pq_kernel<<<96, 256, 0, stream>>>(query, Wq1, b1, Wq2, b2, Wq3, b3, pqb);

  const size_t smem_c = 132096          // A tile  64x1032 bf16
                      + 135168          // B slab 256x264 bf16
                      + 32 * 256 * 4    // pq
                      + 1024 * 4        // normed_v
                      + 64 * 4;         // score accum  => 304,384 B < 320 KB/WGP
  scores_kernel<<<M_TOT / M_TILE, 512, smem_c, stream>>>(value, Wv1, Wv2, Wv3,
                                                         pqb, nv, scores);
  softmax_kernel<<<96, 256, 0, stream>>>(scores, mask, attn);
  combine_kernel<<<M_TOT / 256, 256, 0, stream>>>(attn, ga1, ga2, ga3,
                                                  comb, out + M_TOT, out + 2 * M_TOT);
  context_kernel<<<128, 256, 0, stream>>>(value, comb, out);
}
